// CausalSelfAttention_5282809774464
// MI455X (gfx1250) — matrix operand order repaired and checked by reading, not compiled, not run on hardware
//
#include <hip/hip_runtime.h>
#include <hip/hip_bf16.h>
#include <math.h>

// ---------------------------------------------------------------------------
// CausalSelfAttention for MI455X (gfx1250, wave32, WMMA bf16->f32)
// B=4, T=2048, D=1024, H=16, HD=64
// GEMM staging: A tile via TENSOR_LOAD_TO_LDS (TDM, TENSORcnt),
//               B tile via GLOBAL_LOAD_ASYNC_TO_LDS_B128 (ASYNCcnt),
// double-buffered.
// ---------------------------------------------------------------------------

typedef __bf16 bf16_t;
typedef __attribute__((ext_vector_type(16))) __bf16 v16bf;
typedef __attribute__((ext_vector_type(8)))  float  v8f;
typedef __attribute__((ext_vector_type(4)))  unsigned int u32x4;
typedef __attribute__((ext_vector_type(8)))  int i32x8;
typedef __attribute__((ext_vector_type(4)))  int i32x4;

union BF16Frag {          // one WMMA 16-bit operand: 16 bf16 = 8 VGPRs
  v16bf v;
  u32x4 q[2];             // two 128-bit load chunks
};

__device__ __forceinline__ v8f wmma_bf16(const BF16Frag& a, const BF16Frag& b, v8f c) {
  return __builtin_amdgcn_wmma_f32_16x16x32_bf16(
      /*neg_a=*/false, a.v, /*neg_b=*/false, b.v,
      /*c_mod=*/(short)0, c, /*reuse_a=*/false, /*reuse_b=*/false);
}

// async copy: global (16B per lane) -> LDS, tracked by ASYNCcnt
__device__ __forceinline__ void async_copy_b128(unsigned lds_off, const void* gaddr) {
  asm volatile("global_load_async_to_lds_b128 %0, %1, off"
               :: "v"(lds_off), "v"(gaddr) : "memory");
}

// TDM: 2D tile (tile_w elems of data_size=2B per row, tile_h rows, row pitch
// in elements) global -> LDS, tracked by TENSORcnt. Issued per-wave.
__device__ __forceinline__ void tdm_load_2d(unsigned lds_off, const void* gaddr,
                                            unsigned tile_w, unsigned tile_h,
                                            unsigned pitch_elems) {
  unsigned long long ga = (unsigned long long)(uintptr_t)gaddr;
  u32x4 g0;
  g0[0] = 1u;                                            // count=1, user mode
  g0[1] = lds_off;                                       // lds_addr (bytes)
  g0[2] = (unsigned)(ga & 0xFFFFFFFFu);                  // global_addr[31:0]
  g0[3] = (unsigned)((ga >> 32) & 0x01FFFFFFu) | (2u << 30); // addr[56:32] | type=2
  i32x8 g1;
  g1[0] = (int)(1u << 16);                               // data_size=1 (2 bytes)
  g1[1] = (int)((tile_w & 0xFFFFu) << 16);               // tensor_dim0 lo (==tile_w)
  g1[2] = (int)(((tile_w >> 16) & 0xFFFFu) |
                ((tile_h & 0xFFFFu) << 16));             // td0 hi | tensor_dim1 lo
  g1[3] = (int)(((tile_h >> 16) & 0xFFFFu) |
                ((tile_w & 0xFFFFu) << 16));             // td1 hi | tile_dim0
  g1[4] = (int)(tile_h & 0xFFFFu);                       // tile_dim1 | tile_dim2=0
  g1[5] = (int)pitch_elems;                              // tensor_dim0_stride lo
  g1[6] = 0;                                             // stride hi | d1stride lo
  g1[7] = 0;                                             // d1stride hi
  i32x4 z4 = {0, 0, 0, 0};
#if __clang_major__ >= 23
  i32x8 z8 = {0, 0, 0, 0, 0, 0, 0, 0};
  __builtin_amdgcn_tensor_load_to_lds(g0, g1, z4, z4, z8, 0);
#else
  __builtin_amdgcn_tensor_load_to_lds(g0, g1, z4, z4, 0);
#endif
}

// ds_swizzle xor-shuffle (group-of-32 mode: offset = xor<<10 | and_mask 0x1F)
template <int XORMASK>
__device__ __forceinline__ float swz_xor(float x) {
  int v = __builtin_amdgcn_ds_swizzle(__builtin_bit_cast(int, x), (XORMASK << 10) | 0x1F);
  return __builtin_bit_cast(float, v);
}
__device__ __forceinline__ float redmax16(float x) {
  x = fmaxf(x, swz_xor<1>(x));
  x = fmaxf(x, swz_xor<2>(x));
  x = fmaxf(x, swz_xor<4>(x));
  x = fmaxf(x, swz_xor<8>(x));
  return x;
}
__device__ __forceinline__ float redsum16(float x) {
  x += swz_xor<1>(x);
  x += swz_xor<2>(x);
  x += swz_xor<4>(x);
  x += swz_xor<8>(x);
  return x;
}

// ---------------------------------------------------------------------------
// Conversion kernels
// ---------------------------------------------------------------------------
__global__ void cvt_f32_to_bf16_kernel(const float* __restrict__ in,
                                       bf16_t* __restrict__ out, size_t n) {
  size_t i = (size_t)blockIdx.x * blockDim.x + threadIdx.x;
  if (i < n) out[i] = (bf16_t)in[i];
}

// W [K,N] f32 row-major  ->  Wt [N,K] bf16 row-major (i.e. B column-major)
__global__ void cvt_w_transpose_kernel(const float* __restrict__ W,
                                       bf16_t* __restrict__ Wt,
                                       int Kd, int Nd) {
  size_t i = (size_t)blockIdx.x * blockDim.x + threadIdx.x;   // output index n*Kd+k
  if (i >= (size_t)Kd * Nd) return;
  int k = (int)(i % Kd);
  int n = (int)(i / Kd);
  Wt[i] = (bf16_t)W[(size_t)k * Nd + n];
}

// qkv bf16 [B*T, 3D] -> Vt bf16 [B*H*HD, T]
__global__ void transpose_v_kernel(const bf16_t* __restrict__ qkv,
                                   bf16_t* __restrict__ vt) {
  constexpr int T = 2048, H = 16, HD = 64, THREED = 3072, TWOD = 2048;
  size_t i = (size_t)blockIdx.x * blockDim.x + threadIdx.x;   // over B*H*HD*T
  int t = (int)(i % T);
  size_t r = i / T;
  int d = (int)(r % HD);
  int bh = (int)(r / HD);
  int b = bh / H, h = bh % H;
  vt[i] = qkv[(size_t)(b * T + t) * THREED + TWOD + h * HD + d];
}

// ---------------------------------------------------------------------------
// Tiled WMMA GEMM: C[M,N] = A[M,K](bf16,rowmajor) @ Bt[N,K](bf16) + bias
// block tile 128x128, 8 waves (4m x 2n), wave tile 32x64 (2x4 of 16x16).
// K-step 32, double-buffered: A tile via TDM, B tile via async-to-LDS.
// ---------------------------------------------------------------------------
template <bool OUT_BF16>
__global__ __launch_bounds__(256)
void gemm_bf16_kernel(const bf16_t* __restrict__ A,
                      const bf16_t* __restrict__ Bt,
                      const float* __restrict__ bias,
                      void* __restrict__ Cout,
                      int M, int N, int K) {
  __shared__ bf16_t lA[2][128 * 32];   // 2 x 8 KB
  __shared__ bf16_t lB[2][128 * 32];   // 2 x 8 KB

  const int tid  = threadIdx.x;
  const int lane = tid & 31;
  const int wave = tid >> 5;        // 0..7
  const int wr   = wave >> 1;       // 0..3 -> m offset 32*wr
  const int wc   = wave & 1;        // 0..1 -> n offset 64*wc
  const int m0   = blockIdx.y * 128;
  const int n0   = blockIdx.x * 128;

  const int lrow = lane & 15;
  const int akb  = (lane < 16) ? 0 : 8;    // A frag K base (ISA 16b A layout)
  const int bkb  = (lane < 16) ? 0 : 8;    // B frag K base (same K order as A)

  // B staging chunks owned by this thread: 128x32 = 512 16B-chunks, 2/thread
  const int srow0 = tid >> 2;
  const int soff0 = (tid & 3) << 3;
  const int srow1 = (tid + 256) >> 2;
  const int soff1 = ((tid + 256) & 3) << 3;

  auto stage = [&](int buf, int k0) {
    if (wave == 0) {   // whole 128x32 A tile in one TDM descriptor
      tdm_load_2d((unsigned)(uintptr_t)&lA[buf][0],
                  &A[(size_t)m0 * K + k0], 32u, 128u, (unsigned)K);
    }
    async_copy_b128((unsigned)(uintptr_t)&lB[buf][srow0 * 32 + soff0],
                    &Bt[(size_t)(n0 + srow0) * K + k0 + soff0]);
    async_copy_b128((unsigned)(uintptr_t)&lB[buf][srow1 * 32 + soff1],
                    &Bt[(size_t)(n0 + srow1) * K + k0 + soff1]);
  };

  v8f acc[2][4];
#pragma unroll
  for (int i = 0; i < 2; i++)
#pragma unroll
    for (int j = 0; j < 4; j++) acc[i][j] = (v8f)(0.0f);

  stage(0, 0);                       // prologue: fill buffer 0

  const int nsteps = K >> 5;
  for (int step = 0; step < nsteps; ++step) {
    const int cur = step & 1;
    __syncthreads();                 // all waves done reading buf[cur^1]
    const bool has_next = (step + 1 < nsteps);
    if (has_next) {
      stage(cur ^ 1, (step + 1) << 5);
      if (wave == 0) __builtin_amdgcn_s_wait_tensorcnt(1);   // cur A tile done
      asm volatile("s_wait_asynccnt 0x2" ::: "memory");      // cur B copies done
    } else {
      if (wave == 0) __builtin_amdgcn_s_wait_tensorcnt(0);
      asm volatile("s_wait_asynccnt 0x0" ::: "memory");
    }
    __syncthreads();                 // publish buf[cur] to all waves

    BF16Frag af[2], bfg[4];
#pragma unroll
    for (int i = 0; i < 2; i++) {
      int r = wr * 32 + i * 16 + lrow;
      af[i].q[0] = *(const u32x4*)&lA[cur][r * 32 + akb];
      af[i].q[1] = *(const u32x4*)&lA[cur][r * 32 + akb + 16];
    }
#pragma unroll
    for (int j = 0; j < 4; j++) {
      int r = wc * 64 + j * 16 + lrow;
      bfg[j].q[0] = *(const u32x4*)&lB[cur][r * 32 + bkb];
      bfg[j].q[1] = *(const u32x4*)&lB[cur][r * 32 + bkb + 16];
    }
#pragma unroll
    for (int i = 0; i < 2; i++)
#pragma unroll
      for (int j = 0; j < 4; j++)
        acc[i][j] = wmma_bf16(af[i], bfg[j], acc[i][j]);
  }

  // epilogue: C layout -> row = base + r (+8 for lanes 16..31), col = lane&15
#pragma unroll
  for (int i = 0; i < 2; i++) {
    const int rowbase = m0 + wr * 32 + i * 16 + ((lane < 16) ? 0 : 8);
#pragma unroll
    for (int j = 0; j < 4; j++) {
      const int col = n0 + wc * 64 + j * 16 + lrow;
      const float bv = bias[col];
#pragma unroll
      for (int r = 0; r < 8; r++) {
        float v = acc[i][j][r] + bv;
        size_t idx = (size_t)(rowbase + r) * N + col;
        if (OUT_BF16) ((bf16_t*)Cout)[idx] = (bf16_t)v;
        else          ((float*)Cout)[idx]  = v;
      }
    }
  }
}

// ---------------------------------------------------------------------------
// Flash attention: grid (T/64, B*H); 4 waves, each owns 16 q-rows.
// qkv bf16 [B*T, 3D]; Vt bf16 [B*H*HD, T]; out bf16 [B*T, D]
// ---------------------------------------------------------------------------
__global__ __launch_bounds__(128)
void flash_attn_kernel(const bf16_t* __restrict__ qkv,
                       const bf16_t* __restrict__ vt,
                       bf16_t* __restrict__ out) {
  constexpr int T = 2048, D = 1024, H = 16, HD = 64, THREED = 3072;
  const int qt   = blockIdx.x;          // q tile (64 rows)
  const int bh   = blockIdx.y;
  const int b    = bh / H, h = bh % H;
  const int lane = threadIdx.x & 31;
  const int wave = threadIdx.x >> 5;    // 0..3
  const int lcol = lane & 15;
  const int half8 = (lane < 16) ? 0 : 8;
  const int akb   = (lane < 16) ? 0 : 8;
  const int bkb   = (lane < 16) ? 0 : 8;

  __shared__ bf16_t pbuf[4][16][64];    // per-wave P tile (8 KB)

  const int q0 = qt * 64 + wave * 16;   // this wave's q rows
  const bf16_t* Qb = qkv + (size_t)(b * T) * THREED + h * HD;
  const bf16_t* Kb = Qb + D;
  const bf16_t* Vb = vt + (size_t)bh * HD * T;

  // Q fragment (16 x 64): 2 K-chunks, resident for the whole j loop
  BF16Frag qf[2];
#pragma unroll
  for (int kc = 0; kc < 2; kc++) {
    const bf16_t* p = Qb + (size_t)(q0 + lcol) * THREED + kc * 32 + akb;
    qf[kc].q[0] = *(const u32x4*)(p);
    qf[kc].q[1] = *(const u32x4*)(p + 16);
  }

  v8f o[4];
#pragma unroll
  for (int t = 0; t < 4; t++) o[t] = (v8f)(0.0f);
  float mrw[8], lrw[8];
#pragma unroll
  for (int r = 0; r < 8; r++) { mrw[r] = -INFINITY; lrw[r] = 0.0f; }

  for (int jt = 0; jt <= qt; ++jt) {
    const int j0 = jt * 64;

    // ---- S = Q @ K^T (16 x 64), f32 ----
    v8f s[4];
#pragma unroll
    for (int t = 0; t < 4; t++) s[t] = (v8f)(0.0f);
#pragma unroll
    for (int t = 0; t < 4; t++) {
#pragma unroll
      for (int kc = 0; kc < 2; kc++) {
        BF16Frag kf;
        const bf16_t* p = Kb + (size_t)(j0 + t * 16 + lcol) * THREED + kc * 32 + bkb;
        kf.q[0] = *(const u32x4*)(p);
        kf.q[1] = *(const u32x4*)(p + 16);
        s[t] = wmma_bf16(qf[kc], kf, s[t]);
      }
    }

    // ---- scale + causal mask ----
    const bool diag = (jt == qt);
#pragma unroll
    for (int t = 0; t < 4; t++) {
#pragma unroll
      for (int r = 0; r < 8; r++) {
        float v = s[t][r] * 0.125f;      // 1/sqrt(64)
        if (diag) {
          int q = q0 + r + half8;
          int j = j0 + t * 16 + lcol;
          if (j > q) v = -INFINITY;
        }
        s[t][r] = v;
      }
    }

    // ---- online softmax ----
    float alpha[8];
#pragma unroll
    for (int r = 0; r < 8; r++) {
      float rm = fmaxf(fmaxf(s[0][r], s[1][r]), fmaxf(s[2][r], s[3][r]));
      rm = redmax16(rm);
      float mn = fmaxf(mrw[r], rm);
      alpha[r] = __expf(mrw[r] - mn);    // 0 on first tile (-inf - finite)
      mrw[r] = mn;
    }
    float rsum[8];
#pragma unroll
    for (int r = 0; r < 8; r++) rsum[r] = 0.0f;
#pragma unroll
    for (int t = 0; t < 4; t++) {
#pragma unroll
      for (int r = 0; r < 8; r++) {
        float pv = __expf(s[t][r] - mrw[r]);
        rsum[r] += pv;
        pbuf[wave][r + half8][t * 16 + lcol] = (bf16_t)pv;
      }
    }
#pragma unroll
    for (int r = 0; r < 8; r++) {
      lrw[r] = lrw[r] * alpha[r] + redsum16(rsum[r]);
    }
#pragma unroll
    for (int t = 0; t < 4; t++)
#pragma unroll
      for (int r = 0; r < 8; r++) o[t][r] *= alpha[r];

    __syncthreads();   // P tile visible (C-layout -> A-layout via LDS)

    // ---- O += P @ V ----
#pragma unroll
    for (int t = 0; t < 4; t++) {
#pragma unroll
      for (int kc = 0; kc < 2; kc++) {
        BF16Frag pa, vb;
        const bf16_t* pp = &pbuf[wave][lcol][kc * 32 + akb];
        pa.q[0] = *(const u32x4*)(pp);
        pa.q[1] = *(const u32x4*)(pp + 16);
        const bf16_t* vp = Vb + (size_t)(t * 16 + lcol) * T + j0 + kc * 32 + bkb;
        vb.q[0] = *(const u32x4*)(vp);
        vb.q[1] = *(const u32x4*)(vp + 16);
        o[t] = wmma_bf16(pa, vb, o[t]);
      }
    }
    __syncthreads();   // protect pbuf before next iteration's stores
  }

  // ---- normalize + store (bf16, row-major [B*T, D]) ----
  float inv[8];
#pragma unroll
  for (int r = 0; r < 8; r++) inv[r] = 1.0f / lrw[r];
#pragma unroll
  for (int t = 0; t < 4; t++) {
#pragma unroll
    for (int r = 0; r < 8; r++) {
      int q = q0 + r + half8;
      out[(size_t)(b * T + q) * D + h * HD + t * 16 + lcol] =
          (bf16_t)(o[t][r] * inv[r]);
    }
  }
}

// ---------------------------------------------------------------------------
// Host-side launcher
// ---------------------------------------------------------------------------
extern "C" void kernel_launch(void* const* d_in, const int* in_sizes, int n_in,
                              void* d_out, int out_size, void* d_ws, size_t ws_size,
                              hipStream_t stream) {
  constexpr int B = 4, T = 2048, D = 1024, H = 16;
  constexpr int M = B * T;           // 8192
  constexpr int THREED = 3 * D;      // 3072

  const float* x      = (const float*)d_in[0];
  const float* W_qkv  = (const float*)d_in[1];
  const float* b_qkv  = (const float*)d_in[2];
  const float* W_proj = (const float*)d_in[3];
  const float* b_proj = (const float*)d_in[4];
  float* outp = (float*)d_out;

  char* ws = (char*)d_ws;
  size_t off = 0;
  auto alloc = [&](size_t bytes) { char* p = ws + off; off += (bytes + 255) & ~(size_t)255; return p; };
  bf16_t* x_bf     = (bf16_t*)alloc((size_t)M * D * 2);        // 16 MB
  bf16_t* wqkv_t   = (bf16_t*)alloc((size_t)THREED * D * 2);   //  6 MB
  bf16_t* wproj_t  = (bf16_t*)alloc((size_t)D * D * 2);        //  2 MB
  bf16_t* qkv_bf   = (bf16_t*)alloc((size_t)M * THREED * 2);   // 48 MB
  bf16_t* vt_bf    = (bf16_t*)alloc((size_t)M * D * 2);        // 16 MB
  bf16_t* attn_bf  = (bf16_t*)alloc((size_t)M * D * 2);        // 16 MB
  (void)ws_size; (void)n_in; (void)in_sizes; (void)out_size;

  // 1) convert inputs to bf16 (weights transposed -> column-major B operand)
  {
    size_t n = (size_t)M * D;
    cvt_f32_to_bf16_kernel<<<(n + 255) / 256, 256, 0, stream>>>(x, x_bf, n);
  }
  {
    size_t n = (size_t)D * THREED;
    cvt_w_transpose_kernel<<<(n + 255) / 256, 256, 0, stream>>>(W_qkv, wqkv_t, D, THREED);
  }
  {
    size_t n = (size_t)D * D;
    cvt_w_transpose_kernel<<<(n + 255) / 256, 256, 0, stream>>>(W_proj, wproj_t, D, D);
  }

  // 2) qkv = x @ W_qkv + b_qkv   (bf16 out)
  {
    dim3 grid(THREED / 128, M / 128);
    gemm_bf16_kernel<true><<<grid, 256, 0, stream>>>(x_bf, wqkv_t, b_qkv,
                                                     (void*)qkv_bf, M, THREED, D);
  }

  // 3) V -> Vt [B*H*HD, T]
  {
    size_t n = (size_t)M * D;
    transpose_v_kernel<<<(n + 255) / 256, 256, 0, stream>>>(qkv_bf, vt_bf);
  }

  // 4) flash attention
  {
    dim3 grid(T / 64, B * H);
    flash_attn_kernel<<<grid, 128, 0, stream>>>(qkv_bf, vt_bf, attn_bf);
  }

  // 5) out = attn @ W_proj + b_proj  (f32 out)
  {
    dim3 grid(D / 128, M / 128);
    gemm_bf16_kernel<false><<<grid, 256, 0, stream>>>(attn_bf, wproj_t, b_proj,
                                                      (void*)outp, M, D, D);
  }
}
